// VanillaVectorQuantizer_89361089560823
// MI455X (gfx1250) — matrix operand order repaired and checked by reading, not compiled, not run on hardware
//
#include <hip/hip_runtime.h>

typedef __attribute__((ext_vector_type(16))) _Float16 v16h;
typedef __attribute__((ext_vector_type(8)))  _Float16 v8h;
typedef __attribute__((ext_vector_type(8)))  float    v8f;

#define Bn   32
#define Dd   64
#define HW   4096      // 64*64
#define Kk   512
#define TILE_N 128     // points per workgroup
#define KC2  128       // codebook columns per async-staged LDS chunk (4 chunks)
#define PAD  72        // padded row length in halves (144B = 36 dwords, bank-friendly)

// ---------------- prep: codebook -> f16 transpose + per-column squared norms -------------
__global__ void vq_prep(const float* __restrict__ cb, _Float16* __restrict__ cbT,
                        float* __restrict__ sqe) {
    int k = blockIdx.x * blockDim.x + threadIdx.x;
    if (k >= Kk) return;
    float s = 0.f;
    #pragma unroll
    for (int d = 0; d < Dd; ++d) {
        float v = cb[d * Kk + k];
        s += v * v;
        cbT[k * Dd + d] = (_Float16)v;
    }
    sqe[k] = s;
}

// --- CDNA5 async global->LDS copy (ASYNCcnt-tracked), bypasses VGPRs ---
__device__ __forceinline__ void async_copy_b128(unsigned lds_off,
                                                unsigned long long gbase,
                                                unsigned goff) {
    asm volatile("global_load_async_to_lds_b128 %0, %1, %2 offset:0"
                 :: "v"(lds_off), "v"(goff), "s"(gbase)
                 : "memory");
}
__device__ __forceinline__ void wait_async0() {
    asm volatile("s_wait_asynccnt 0x0" ::: "memory");
}

// ---------------- main: distances via WMMA, argmin, gather ------------------------------
__global__ __launch_bounds__(256) void vq_main(const float* __restrict__ enc,
                                               const float* __restrict__ cb,
                                               const _Float16* __restrict__ cbT,
                                               const float* __restrict__ sqe,
                                               float* __restrict__ out) {
    __shared__ __align__(16) _Float16 Blds[2][KC2][PAD];  // 36864 B, double-buffered
    __shared__ float sqe_lds[Kk];                         //  2048 B
    __shared__ int   idx_lds[TILE_N];                     //   512 B

    const int tid = threadIdx.x;
    const int n0  = blockIdx.x * TILE_N;       // first point of this tile
    const int b   = n0 / HW;
    const int rem = n0 % HW;                   // multiple of 128 -> stays inside one image

    const int lane = tid & 31;
    const int wave = tid >> 5;          // 0..7, each owns 16 rows
    const int r0   = wave * 16;
    const int m    = lane & 15;         // A row / B col / C col within tile
    const int hi   = lane >> 4;         // half-wave select

    // ---- issue async stage of codebook chunk 0 (128 cols x 64 halves = 16 KB) ----
    #pragma unroll
    for (int i = 0; i < 4; ++i) {
        int flat = i * 256 + tid;                 // 16-byte unit, 1024 total
        int row  = flat >> 3;
        int seg  = flat & 7;
        async_copy_b128((unsigned)(unsigned long long)&Blds[0][row][seg * 8],
                        (unsigned long long)cbT,
                        (unsigned)((0 * KC2 + row) * (Dd * 2) + seg * 16));
    }

    // ---- overlap: fill sqe + assemble A fragments straight from global (f32->f16) ----
    for (int i = tid; i < Kk; i += 256) sqe_lds[i] = sqe[i];

    union AU { v16h v; _Float16 e[16]; } af[2];
    #pragma unroll
    for (int c = 0; c < 2; ++c) {
        #pragma unroll
        for (int j = 0; j < 16; ++j) {
            // A 16x32 f16 fragment: halves 0..7 -> K = c*32+hi*8+j ; 8..15 -> +16
            int d = c * 32 + hi * 8 + ((j < 8) ? j : (j + 8));
            af[c].e[j] = (_Float16)enc[(size_t)(b * Dd + d) * HW + rem + r0 + m];
        }
    }

    float minval[8];
    int   minidx[8];
    #pragma unroll
    for (int r = 0; r < 8; ++r) { minval[r] = 3.4e38f; minidx[r] = 0; }

    for (int kc = 0; kc < Kk / KC2; ++kc) {       // 4 chunks
        wait_async0();          // my async writes for chunk kc have landed in LDS
        __syncthreads();        // everyone's writes visible; prev buffer fully consumed

        if (kc < Kk / KC2 - 1) {                  // prefetch next chunk into other buffer
            #pragma unroll
            for (int i = 0; i < 4; ++i) {
                int flat = i * 256 + tid;
                int row  = flat >> 3;
                int seg  = flat & 7;
                async_copy_b128(
                    (unsigned)(unsigned long long)&Blds[(kc + 1) & 1][row][seg * 8],
                    (unsigned long long)cbT,
                    (unsigned)(((kc + 1) * KC2 + row) * (Dd * 2) + seg * 16));
            }
        }

        const int buf = kc & 1;
        for (int t = 0; t < KC2 / 16; ++t) {      // 8 column tiles
            v8f acc = {};
            #pragma unroll
            for (int c = 0; c < 2; ++c) {         // D = 64 = 2 x K32
                union { v16h v; v8h h[2]; } bm;
                // B 32x16 f16 fragment: halves 0..7 -> K = hi*8+j ; 8..15 -> +16
                bm.h[0] = *(const v8h*)&Blds[buf][t * 16 + m][c * 32 + hi * 8];
                bm.h[1] = *(const v8h*)&Blds[buf][t * 16 + m][c * 32 + 16 + hi * 8];
                acc = __builtin_amdgcn_wmma_f32_16x16x32_f16(
                    false, af[c].v, false, bm.v, (short)0, acc, false, false);
            }
            // dist = ||e||^2 - 2*x.e  (||x||^2 irrelevant for argmin)
            int col = kc * KC2 + t * 16 + m;
            float se = sqe_lds[col];
            #pragma unroll
            for (int r = 0; r < 8; ++r) {
                float dist = se - 2.0f * acc[r];
                if (dist < minval[r]) { minval[r] = dist; minidx[r] = col; }
            }
        }
    }

    // ---- argmin across the 16 columns held in lanes of each half-wave ----
    #pragma unroll
    for (int r = 0; r < 8; ++r) {
        float v = minval[r];
        int  ix = minidx[r];
        #pragma unroll
        for (int mask = 1; mask < 16; mask <<= 1) {
            float ov = __shfl_xor(v, mask, 32);
            int   oi = __shfl_xor(ix, mask, 32);
            if (ov < v || (ov == v && oi < ix)) { v = ov; ix = oi; }
        }
        minval[r] = v; minidx[r] = ix;
    }
    if (m == 0) {
        // lanes 0..15 half (hi=0): rows r0+0..7 ; lanes 16..31 (hi=1): rows r0+8..15
        #pragma unroll
        for (int r = 0; r < 8; ++r) idx_lds[r0 + hi * 8 + r] = minidx[r];
    }
    __syncthreads();

    // ---- gather winning codebook columns (f32) and write [B,D,H,W] output ----
    #pragma unroll
    for (int i = 0; i < (TILE_N * Dd) / 256; ++i) {
        int flat = i * 256 + tid;
        int d  = flat >> 7;
        int nl = flat & 127;
        out[(size_t)(b * Dd + d) * HW + rem + nl] = cb[d * Kk + idx_lds[nl]];
    }
}

extern "C" void kernel_launch(void* const* d_in, const int* in_sizes, int n_in,
                              void* d_out, int out_size, void* d_ws, size_t ws_size,
                              hipStream_t stream) {
    const float* enc = (const float*)d_in[0];   // [32, 64, 64, 64] f32
    const float* cb  = (const float*)d_in[1];   // [64, 512] f32
    float* out = (float*)d_out;                 // [32, 64, 64, 64] f32

    _Float16* cbT = (_Float16*)d_ws;                              // 512*64*2 = 64 KB
    float*    sqe = (float*)((char*)d_ws + (size_t)Kk * Dd * 2);  // 2 KB

    vq_prep<<<(Kk + 255) / 256, 256, 0, stream>>>(cb, cbT, sqe);

    int nblocks = (Bn * HW) / TILE_N;   // 1024
    vq_main<<<nblocks, 256, 0, stream>>>(enc, cb, cbT, sqe, out);
}
